// SCNCore_36936718745871
// MI455X (gfx1250) — compile-verified, matrix operand order checked by reading, not run on hardware
//
#include <hip/hip_runtime.h>
#include <hip/hip_bf16.h>
#include <math.h>

// ---------------------------------------------------------------------------
// SCN concept-gated LSTM cell, CDNA5 (gfx1250) WMMA implementation.
//   hidden = sigmoid( (concept@Wci^T) * (x@Win^T) + (concept@Wcs^T) * (h@Wst^T) )
//   i,f,o,cc = split(hidden, 4);  c_t = i*cc + f*c_state;  h_t = o*tanh(c_t)
// B = 1024, IN = H = 1024, C = 1000 (padded to 1024), 4H = 4096.
//
// Software-pipelined: double-buffered LDS + register prefetch of the next
// K-step's global data so HBM latency overlaps the v_wmma work.
// ---------------------------------------------------------------------------

typedef __attribute__((ext_vector_type(16))) __bf16 v16bf;
typedef __attribute__((ext_vector_type(8)))  __bf16 v8bf;
typedef __attribute__((ext_vector_type(8)))  float  v8f;

#define BDIM   1024
#define KDIM   1024   // padded K (IN == H == 1024; C == 1000 padded to 1024)
#define CDIM   1000
#define N4H    4096
#define BM     64     // block tile M
#define BN     64     // block tile N
#define BK     32     // K step (bf16 WMMA K)
#define NITER  (KDIM / BK)
#define NMAT   7      // Ax, Ah, Ac, Win, Wst, Wci, Wcs
#define TILE_E (BM * BK)   // 2048 bf16 per tile

// Global -> registers: one 8-float group per thread per matrix.
// Groups of 8 columns are either fully inside [0, klimit) or fully padded
// (klimit is 1024 or 1000; both are multiples of 8, rows are 16B aligned).
__device__ __forceinline__ void gload_group(const float* __restrict__ src,
                                            int grow_base, int ldk, int k0,
                                            int klimit, int tid,
                                            float4& lo, float4& hi) {
    const int row  = tid >> 2;          // 0..63
    const int cg   = (tid & 3) << 3;    // 0,8,16,24
    const int gcol = k0 + cg;
    if (gcol + 8 <= klimit) {
        const float4* p =
            reinterpret_cast<const float4*>(src + (size_t)(grow_base + row) * ldk + gcol);
        lo = p[0];
        hi = p[1];
    } else {
        lo = make_float4(0.f, 0.f, 0.f, 0.f);
        hi = lo;
    }
}

// Registers -> bf16 LDS tile (row-major 64x32).
__device__ __forceinline__ void cvt_store(__bf16* __restrict__ dst,
                                          const float4& lo, const float4& hi,
                                          int tid) {
    const int row = tid >> 2;
    const int cg  = (tid & 3) << 3;
    v8bf v;
    v[0] = (__bf16)lo.x; v[1] = (__bf16)lo.y; v[2] = (__bf16)lo.z; v[3] = (__bf16)lo.w;
    v[4] = (__bf16)hi.x; v[5] = (__bf16)hi.y; v[6] = (__bf16)hi.z; v[7] = (__bf16)hi.w;
    *reinterpret_cast<v8bf*>(dst + row * BK + cg) = v;
}

// Load a 16x32 bf16 WMMA operand fragment from a 64x32 LDS tile.
// lane r = matrix row (A: M, B: N); upper lane half holds upper K half.
// Contiguous-per-lane load => same K permutation for A and B (sum invariant).
__device__ __forceinline__ v16bf load_frag(const __bf16* __restrict__ s,
                                           int row_base, int lane) {
    const int r  = lane & 15;
    const int hk = lane >> 4;
    return *reinterpret_cast<const v16bf*>(s + (row_base + r) * BK + hk * 16);
}

__global__ void __launch_bounds__(256)
scn_gates_wmma(const float* __restrict__ X,     // core_input [B, IN]
               const float* __restrict__ Hst,   // h_state    [B, H]
               const float* __restrict__ Cpt,   // concept    [B, C]
               const float* __restrict__ Win,   // [4H, IN]
               const float* __restrict__ Wst,   // [4H, H]
               const float* __restrict__ Wci,   // [4H, C]
               const float* __restrict__ Wcs,   // [4H, C]
               float* __restrict__ hidden) {    // [B, 4H] ws
    // Double-buffered staging: [buf][matrix][64*32 bf16] = 2 * 28 KB.
    __shared__ __align__(64) __bf16 smem[2][NMAT][TILE_E];

    const int tid  = threadIdx.x;
    const int wave = tid >> 5;          // 0..7
    const int lane = tid & 31;
    const int wm   = wave >> 1;         // 0..3 : wave M tile within block
    const int wn   = wave & 1;          // 0..1 : wave N pair within block

    const int m0 = blockIdx.y * BM;     // output rows [m0, m0+64)
    const int n0 = blockIdx.x * BN;     // output cols [n0, n0+64)

    const int arow  = wm * 16;          // wave's A rows inside LDS tile
    const int brow0 = wn * 32;          // wave's first B tile inside LDS tile

    v8f acc_xi[2] = {{}, {}};
    v8f acc_hs[2] = {{}, {}};
    v8f acc_gi[2] = {{}, {}};
    v8f acc_gs[2] = {{}, {}};

    float4 lo[NMAT], hi[NMAT];

    // Prologue: fill buffer 0 with K-step 0.
    gload_group(X,   m0, KDIM, 0, KDIM, tid, lo[0], hi[0]);
    gload_group(Hst, m0, KDIM, 0, KDIM, tid, lo[1], hi[1]);
    gload_group(Cpt, m0, CDIM, 0, CDIM, tid, lo[2], hi[2]);
    gload_group(Win, n0, KDIM, 0, KDIM, tid, lo[3], hi[3]);
    gload_group(Wst, n0, KDIM, 0, KDIM, tid, lo[4], hi[4]);
    gload_group(Wci, n0, CDIM, 0, CDIM, tid, lo[5], hi[5]);
    gload_group(Wcs, n0, CDIM, 0, CDIM, tid, lo[6], hi[6]);
#pragma unroll
    for (int m = 0; m < NMAT; ++m) cvt_store(smem[0][m], lo[m], hi[m], tid);
    __syncthreads();

    for (int it = 0; it < NITER; ++it) {
        const int cur = it & 1;
        const __bf16* sb = &smem[cur][0][0];

        // Issue next K-step's global loads BEFORE the matrix work so the
        // HBM/L2 latency overlaps the WMMAs.
        if (it + 1 < NITER) {
            const int kn = (it + 1) * BK;
            gload_group(X,   m0, KDIM, kn, KDIM, tid, lo[0], hi[0]);
            gload_group(Hst, m0, KDIM, kn, KDIM, tid, lo[1], hi[1]);
            gload_group(Cpt, m0, CDIM, kn, CDIM, tid, lo[2], hi[2]);
            gload_group(Win, n0, KDIM, kn, KDIM, tid, lo[3], hi[3]);
            gload_group(Wst, n0, KDIM, kn, KDIM, tid, lo[4], hi[4]);
            gload_group(Wci, n0, CDIM, kn, CDIM, tid, lo[5], hi[5]);
            gload_group(Wcs, n0, CDIM, kn, CDIM, tid, lo[6], hi[6]);
        }

        // Matrix work on the current buffer: 8 independent WMMAs (no
        // WMMA->WMMA RAW hazards, distinct accumulators).
        const v16bf ax = load_frag(sb + 0 * TILE_E, arow, lane);
        const v16bf ah = load_frag(sb + 1 * TILE_E, arow, lane);
        const v16bf ac = load_frag(sb + 2 * TILE_E, arow, lane);

#pragma unroll
        for (int j = 0; j < 2; ++j) {
            const v16bf b_in = load_frag(sb + 3 * TILE_E, brow0 + j * 16, lane);
            const v16bf b_st = load_frag(sb + 4 * TILE_E, brow0 + j * 16, lane);
            const v16bf b_ci = load_frag(sb + 5 * TILE_E, brow0 + j * 16, lane);
            const v16bf b_cs = load_frag(sb + 6 * TILE_E, brow0 + j * 16, lane);

            acc_xi[j] = __builtin_amdgcn_wmma_f32_16x16x32_bf16(
                false, ax, false, b_in, (short)0, acc_xi[j], false, false);
            acc_hs[j] = __builtin_amdgcn_wmma_f32_16x16x32_bf16(
                false, ah, false, b_st, (short)0, acc_hs[j], false, false);
            acc_gi[j] = __builtin_amdgcn_wmma_f32_16x16x32_bf16(
                false, ac, false, b_ci, (short)0, acc_gi[j], false, false);
            acc_gs[j] = __builtin_amdgcn_wmma_f32_16x16x32_bf16(
                false, ac, false, b_cs, (short)0, acc_gs[j], false, false);
        }

        // Drain prefetch into the other buffer and hand it over.
        if (it + 1 < NITER) {
            __bf16* sn = &smem[1 - cur][0][0];
#pragma unroll
            for (int m = 0; m < NMAT; ++m)
                cvt_store(sn + m * TILE_E, lo[m], hi[m], tid);
            __syncthreads();
        }
    }

    // Epilogue: sigmoid(gi*xi + gs*hs) -> hidden_vec.
    // C/D layout: VGPR e, lanes 0-15 -> M = e, N = lane; lanes 16-31 -> M = e+8.
    const int rbase = m0 + wm * 16 + ((lane >> 4) << 3);
    const int cbase = n0 + wn * 32 + (lane & 15);
#pragma unroll
    for (int j = 0; j < 2; ++j) {
#pragma unroll
        for (int e = 0; e < 8; ++e) {
            const float z  = acc_gi[j][e] * acc_xi[j][e] + acc_gs[j][e] * acc_hs[j][e];
            const float sv = 1.0f / (1.0f + __expf(-z));
            hidden[(size_t)(rbase + e) * N4H + (cbase + j * 16)] = sv;
        }
    }
}

__global__ void __launch_bounds__(256)
scn_combine(const float* __restrict__ hidden,   // [B, 4H]
            const float* __restrict__ c_state,  // [B, H]
            float* __restrict__ out) {          // h_t [B,H] then c_t [B,H]
    const int idx = blockIdx.x * 256 + threadIdx.x;   // 0 .. B*H-1
    const int b = idx >> 10;
    const int n = idx & 1023;
    const float* hv = hidden + (size_t)b * N4H;
    const float i_t = hv[n];
    const float f_t = hv[n + 1024];
    const float o_t = hv[n + 2048];
    const float cc  = hv[n + 3072];
    const float c_t = i_t * cc + f_t * c_state[idx];
    const float h_t = o_t * tanhf(c_t);
    out[idx] = h_t;
    out[idx + BDIM * 1024] = c_t;
}

extern "C" void kernel_launch(void* const* d_in, const int* in_sizes, int n_in,
                              void* d_out, int out_size, void* d_ws, size_t ws_size,
                              hipStream_t stream) {
    (void)in_sizes; (void)n_in; (void)out_size; (void)ws_size;
    const float* X   = (const float*)d_in[0];   // core_input [1024,1024]
    const float* Hs  = (const float*)d_in[1];   // h_state    [1024,1024]
    const float* Cs  = (const float*)d_in[2];   // c_state    [1024,1024]
    const float* Cpt = (const float*)d_in[3];   // concept    [1024,1000]
    const float* Win = (const float*)d_in[4];   // [4096,1024]
    const float* Wst = (const float*)d_in[5];   // [4096,1024]
    const float* Wci = (const float*)d_in[6];   // [4096,1000]
    const float* Wcs = (const float*)d_in[7];   // [4096,1000]

    float* hidden = (float*)d_ws;               // [1024,4096] = 16 MB scratch

    dim3 grid1(N4H / BN, BDIM / BM);            // (64, 16)
    scn_gates_wmma<<<grid1, dim3(256), 0, stream>>>(X, Hs, Cpt, Win, Wst, Wci, Wcs,
                                                    hidden);

    scn_combine<<<dim3((BDIM * 1024) / 256), dim3(256), 0, stream>>>(hidden, Cs,
                                                                     (float*)d_out);
}
